// GLCA_38577396253243
// MI455X (gfx1250) — matrix operand order repaired and checked by reading, not compiled, not run on hardware
//
#include <hip/hip_runtime.h>
#include <stdint.h>

// ---------------- problem dims ----------------
#define DB 4
#define DC 512
#define DNCLS 19
#define DT 256
#define DH 128
#define DW 128
#define DHW 16384
#define DPH 16
#define DPW 16
#define DNP 256    // B*8*8 patches
#define DPHW 256   // 16*16 pixels per patch

typedef __bf16 bf16_t;
typedef bf16_t bf16x16 __attribute__((ext_vector_type(16)));
typedef float f32x8 __attribute__((ext_vector_type(8)));

__device__ __forceinline__ float bf2f(unsigned short u) {
  return __uint_as_float(((unsigned)u) << 16);
}
__device__ __forceinline__ unsigned short f2bf(float f) {
  unsigned u = __float_as_uint(f);
  u += 0x7FFFu + ((u >> 16) & 1u);   // round-to-nearest-even
  return (unsigned short)(u >> 16);
}

union FragBF { bf16x16 v; uint4 q[2]; };

// ---- gfx1250 async global->LDS (ASYNCcnt-tracked), ISA 15.18.3 op 98 ----
// Each active lane copies 16 bytes from its global address to its LDS address.
__device__ __forceinline__ void async_load_b128_to_lds(unsigned lds_byte_addr,
                                                       const unsigned short* gaddr) {
  asm volatile("global_load_async_to_lds_b128 %0, %1, off"
               :: "v"(lds_byte_addr), "v"(gaddr)
               : "memory");
}
__device__ __forceinline__ void wait_async0() {
#if __has_builtin(__builtin_amdgcn_s_wait_asynccnt)
  __builtin_amdgcn_s_wait_asynccnt(0);
#else
  asm volatile("s_wait_asynccnt 0x0" ::: "memory");
#endif
}
__device__ __forceinline__ unsigned lds_addr_of(const void* p) {
  return (unsigned)(unsigned long long)p;   // low 32 bits of flat LDS aperture = LDS offset
}

// =====================================================================
// Generic batched WMMA GEMM:  Out[b] = epi( A[b](MxK) * B[b](KxN) )
// 256 threads = 8 waves (4 M x 2 N) -> 64x64 tile, 2 WMMAs/wave/K-step
// (A-fragment reused across both B fragments).
// A tile staged with GLOBAL_LOAD_ASYNC_TO_LDS_B128 when the M-slab is full.
// BMODE: 0 = dense B (row stride N), 1 = patch view of (B,C,H,W) bf16
// OMODE: 0 = f32 dense, 1 = bf16 dense, 2 = bf16 transposed (n*M+m),
//        3 = bf16 patch-scatter into concat tensor (chanOff + patch_recover)
// EPI:   0 none, 1 +bias[m], 2 relu(x*s[m]+b[m]), 3 relu6(x*s[m]+b[m]),
//        4 relu6(x + res[b,m,n])
// Assumes K%32==0 and N%64==0 (true at every call site).
// =====================================================================
template<int BMODE, int OMODE, int EPI>
__global__ __launch_bounds__(256) void wmma_gemm_kernel(
    const unsigned short* __restrict__ A, long long strideA,
    const unsigned short* __restrict__ Bsrc, long long strideB,
    void* __restrict__ Out, long long strideOut,
    int M, int N, int K, int Mreal,
    const float* __restrict__ ep0, const float* __restrict__ ep1,
    long long epStride, int chanOff)
{
  constexpr int LDA = 40, LDB = 40;         // padded ushort row strides
  __shared__ unsigned short As[64 * LDA];   // 64 m x 32 k
  __shared__ unsigned short Bs[64 * LDB];   // transposed: 64 n x 32 k

  const int tid   = threadIdx.x;
  const int batch = blockIdx.z;
  const int mBase = blockIdx.y * 64;
  const int nBase = blockIdx.x * 64;

  const unsigned short* Ab = A + (long long)batch * strideA;
  const unsigned short* Bb = Bsrc + ((BMODE == 0) ? (long long)batch * strideB : 0ll);

  // patch decomposition (used by BMODE==1 and OMODE==3)
  const int pb = batch >> 6;
  const int ph = (batch >> 3) & 7;
  const int pw = batch & 7;

  // A tile loader: 64 rows x 32 k, one uint4 (8 ushorts) per thread
  const int ar  = tid >> 2;
  const int aks = (tid & 3) * 8;
  // B tile loader: 32 k x 64 n, 8 n per thread at one k
  const int bk  = tid >> 3;
  const int bn8 = (tid & 7) * 8;

  const bool fullM = (mBase + 64 <= M);     // block-uniform
  const unsigned asA = lds_addr_of(&As[ar * LDA + aks]);

  f32x8 acc0 = {}, acc1 = {};

  const int lane  = tid & 31;
  const int wid   = tid >> 5;
  const int waveM = wid >> 1;   // 0..3
  const int waveN = wid & 1;    // 0..1
  const int l16   = lane & 15;
  const int hk    = lane >> 4;  // half-K select per ISA bf16 layout

  for (int k0 = 0; k0 < K; k0 += 32) {
    // ---- A tile: async DMA into LDS (full slab) or guarded sync fill ----
    if (fullM) {
      async_load_b128_to_lds(asA, Ab + (size_t)(mBase + ar) * K + k0 + aks);
    } else {
      int gm = mBase + ar;
      uint4 val = {0u, 0u, 0u, 0u};
      if (gm < M) val = *reinterpret_cast<const uint4*>(Ab + (size_t)gm * K + k0 + aks);
      *reinterpret_cast<uint4*>(&As[ar * LDA + aks]) = val;
    }
    // ---- B tile (transposed into LDS) ----
    {
      int gk = k0 + bk;
      unsigned short e[8];
      if (BMODE == 0) {
        uint4 raw = *reinterpret_cast<const uint4*>(Bb + (size_t)gk * N + nBase + bn8);
        e[0] = (unsigned short)(raw.x & 0xffffu); e[1] = (unsigned short)(raw.x >> 16);
        e[2] = (unsigned short)(raw.y & 0xffffu); e[3] = (unsigned short)(raw.y >> 16);
        e[4] = (unsigned short)(raw.z & 0xffffu); e[5] = (unsigned short)(raw.z >> 16);
        e[6] = (unsigned short)(raw.w & 0xffffu); e[7] = (unsigned short)(raw.w >> 16);
        if (k0 + 32 < K)  // prefetch next K-step (global_prefetch_b8)
          __builtin_prefetch(Bb + (size_t)(gk + 32) * N + nBase + bn8, 0, 3);
      } else {
        // k -> channel, n -> patch pixel inside (B,C,H,W) bf16 tensor
        const unsigned short* base =
            Bb + (((size_t)pb * DC + gk) * DH + ph * DPH) * DW + pw * DPW;
        #pragma unroll
        for (int j = 0; j < 8; ++j) {
          int p = nBase + bn8 + j;
          e[j] = base[(p >> 4) * DW + (p & 15)];
        }
      }
      #pragma unroll
      for (int j = 0; j < 8; ++j) Bs[(bn8 + j) * LDB + bk] = e[j];
    }
    wait_async0();
    __syncthreads();

    FragBF fa, fb0, fb1;
    const unsigned short* arow = &As[(waveM * 16 + l16) * LDA];
    fa.q[0] = *reinterpret_cast<const uint4*>(arow + hk * 8);        // K 0..7 / 8..15
    fa.q[1] = *reinterpret_cast<const uint4*>(arow + 16 + hk * 8);   // K 16..23 / 24..31
    const unsigned short* brow0 = &Bs[(waveN * 32 + l16) * LDB];
    const unsigned short* brow1 = &Bs[(waveN * 32 + 16 + l16) * LDB];
    fb0.q[0] = *reinterpret_cast<const uint4*>(brow0 + hk * 8);      // K 0..7 / 8..15
    fb0.q[1] = *reinterpret_cast<const uint4*>(brow0 + 16 + hk * 8);
    fb1.q[0] = *reinterpret_cast<const uint4*>(brow1 + hk * 8);
    fb1.q[1] = *reinterpret_cast<const uint4*>(brow1 + 16 + hk * 8);

    acc0 = __builtin_amdgcn_wmma_f32_16x16x32_bf16(false, fa.v, false, fb0.v,
                                                   (short)0, acc0, false, false);
    acc1 = __builtin_amdgcn_wmma_f32_16x16x32_bf16(false, fa.v, false, fb1.v,
                                                   (short)0, acc1, false, false);
    __syncthreads();
  }

  // epilogue + store (C/D layout: VGPR r -> row r (+8 upper lanes), col = lane&15)
  #pragma unroll
  for (int nf = 0; nf < 2; ++nf) {
    const f32x8& acc = nf ? acc1 : acc0;
    const int gn = nBase + waveN * 32 + nf * 16 + l16;
    #pragma unroll
    for (int r = 0; r < 8; ++r) {
      int gm = mBase + waveM * 16 + r + hk * 8;
      if (gm < M && gn < N) {
        float y = acc[r];
        if (EPI == 1) {
          y += (gm < Mreal) ? ep0[gm] : 0.0f;
        } else if (EPI == 2) {
          y = fmaxf(fmaf(y, ep0[gm], ep1[gm]), 0.0f);
        } else if (EPI == 3) {
          y = fminf(fmaxf(fmaf(y, ep0[gm], ep1[gm]), 0.0f), 6.0f);
        } else if (EPI == 4) {
          float res = ep0[(long long)batch * epStride + (size_t)gm * N + gn];
          y = fminf(fmaxf(y + res, 0.0f), 6.0f);
        }
        if (OMODE == 0) {
          reinterpret_cast<float*>(Out)[(long long)batch * strideOut + (size_t)gm * N + gn] = y;
        } else if (OMODE == 1) {
          reinterpret_cast<unsigned short*>(Out)[(long long)batch * strideOut + (size_t)gm * N + gn] = f2bf(y);
        } else if (OMODE == 2) {
          reinterpret_cast<unsigned short*>(Out)[(long long)batch * strideOut + (size_t)gn * M + gm] = f2bf(y);
        } else { // OMODE == 3: patch_recover + channel offset into (B, 2C, H, W)
          int py = gn >> 4, px = gn & 15;
          size_t o = (((size_t)pb * (2 * DC) + chanOff + gm) * DH + ph * DPH + py) * DW
                     + pw * DPW + px;
          reinterpret_cast<unsigned short*>(Out)[o] = f2bf(y);
        }
      }
    }
  }
}

// =====================================================================
// Implicit-GEMM 3x3 SAME conv via WMMA. K = IC*9, im2col in B loader.
// Weights staged with async global->LDS DMA (OC is always a multiple of 64).
// Epilogue: relu6(x*scale[o] + shift[o]).
// =====================================================================
template<bool OUT_BF16>
__global__ __launch_bounds__(256) void wmma_conv3x3_kernel(
    const unsigned short* __restrict__ Wt,      // (OC, IC*9) bf16
    const unsigned short* __restrict__ Xin,     // (B, IC, H, W) bf16
    int IC,
    const float* __restrict__ scale, const float* __restrict__ shift,
    void* __restrict__ Out, int OC)
{
  constexpr int LDA = 40, LDB = 40;
  __shared__ unsigned short As[64 * LDA];
  __shared__ unsigned short Bs[64 * LDB];

  const int tid   = threadIdx.x;
  const int batch = blockIdx.z;
  const int mBase = blockIdx.y * 64;
  const int nBase = blockIdx.x * 64;
  const int K = IC * 9;

  const int ar  = tid >> 2;
  const int aks = (tid & 3) * 8;
  const int bk  = tid >> 3;
  const int bn8 = (tid & 7) * 8;
  const unsigned asA = lds_addr_of(&As[ar * LDA + aks]);

  f32x8 acc0 = {}, acc1 = {};
  const int lane  = tid & 31;
  const int wid   = tid >> 5;
  const int waveM = wid >> 1;
  const int waveN = wid & 1;
  const int l16   = lane & 15;
  const int hk    = lane >> 4;

  for (int k0 = 0; k0 < K; k0 += 32) {
    // weights tile via async DMA to LDS
    async_load_b128_to_lds(asA, Wt + (size_t)(mBase + ar) * K + k0 + aks);
    { // im2col B tile
      int gk = k0 + bk;
      int c  = gk / 9;
      int rr = gk - c * 9;
      int r3 = rr / 3;
      int ky = r3 - 1;
      int kx = (rr - r3 * 3) - 1;
      const unsigned short* base = Xin + ((size_t)batch * IC + c) * DHW;
      #pragma unroll
      for (int j = 0; j < 8; ++j) {
        int n = nBase + bn8 + j;
        int y = (n >> 7) + ky;
        int x = (n & 127) + kx;
        unsigned short e = 0;
        if ((unsigned)y < (unsigned)DH && (unsigned)x < (unsigned)DW) e = base[y * DW + x];
        Bs[(bn8 + j) * LDB + bk] = e;
      }
      if (k0 + 32 < K) { // prefetch next K-step's input row (always a valid address)
        int gk2 = gk + 32;
        int c2 = gk2 / 9;
        const unsigned short* base2 = Xin + ((size_t)batch * IC + c2) * DHW;
        __builtin_prefetch(base2 + (((nBase + bn8) >> 7)) * DW + ((nBase + bn8) & 127), 0, 3);
      }
    }
    wait_async0();
    __syncthreads();

    FragBF fa, fb0, fb1;
    const unsigned short* arow = &As[(waveM * 16 + l16) * LDA];
    fa.q[0] = *reinterpret_cast<const uint4*>(arow + hk * 8);
    fa.q[1] = *reinterpret_cast<const uint4*>(arow + 16 + hk * 8);
    const unsigned short* brow0 = &Bs[(waveN * 32 + l16) * LDB];
    const unsigned short* brow1 = &Bs[(waveN * 32 + 16 + l16) * LDB];
    fb0.q[0] = *reinterpret_cast<const uint4*>(brow0 + hk * 8);
    fb0.q[1] = *reinterpret_cast<const uint4*>(brow0 + 16 + hk * 8);
    fb1.q[0] = *reinterpret_cast<const uint4*>(brow1 + hk * 8);
    fb1.q[1] = *reinterpret_cast<const uint4*>(brow1 + 16 + hk * 8);

    acc0 = __builtin_amdgcn_wmma_f32_16x16x32_bf16(false, fa.v, false, fb0.v,
                                                   (short)0, acc0, false, false);
    acc1 = __builtin_amdgcn_wmma_f32_16x16x32_bf16(false, fa.v, false, fb1.v,
                                                   (short)0, acc1, false, false);
    __syncthreads();
  }

  #pragma unroll
  for (int nf = 0; nf < 2; ++nf) {
    const f32x8& acc = nf ? acc1 : acc0;
    const int gn = nBase + waveN * 32 + nf * 16 + l16;
    #pragma unroll
    for (int r = 0; r < 8; ++r) {
      int gm = mBase + waveM * 16 + r + hk * 8;
      if (gm < OC) {
        float y = fminf(fmaxf(fmaf(acc[r], scale[gm], shift[gm]), 0.0f), 6.0f);
        size_t o = ((size_t)batch * OC + gm) * DHW + gn;
        if (OUT_BF16) reinterpret_cast<unsigned short*>(Out)[o] = f2bf(y);
        else          reinterpret_cast<float*>(Out)[o] = y;
      }
    }
  }
}

// ---------------- small helper kernels ----------------
__global__ void k_f32_to_bf16(const float* __restrict__ src,
                              unsigned short* __restrict__ dst, long long count) {
  long long i = (long long)blockIdx.x * blockDim.x + threadIdx.x;
  if (i < count) dst[i] = f2bf(src[i]);
}

__global__ void k_convert_pad_rows(const float* __restrict__ src,
                                   unsigned short* __restrict__ dst,
                                   int rowsReal, int rowsPad, int cols) {
  long long i = (long long)blockIdx.x * blockDim.x + threadIdx.x;
  long long total = (long long)rowsPad * cols;
  if (i < total) {
    int r = (int)(i / cols);
    dst[i] = (r < rowsReal) ? f2bf(src[i]) : (unsigned short)0;
  }
}

// feat -> bf16 copy (dense) AND residual half of the concat tensor
__global__ void k_feat_convert(const float* __restrict__ feat,
                               unsigned short* __restrict__ featbf,
                               unsigned short* __restrict__ xcat) {
  long long i = (long long)blockIdx.x * blockDim.x + threadIdx.x;
  const long long total = (long long)DB * DC * DHW;
  if (i < total) {
    unsigned short v = f2bf(feat[i]);
    featbf[i] = v;
    long long b = i / ((long long)DC * DHW);
    long long rem = i - b * (long long)DC * DHW;
    xcat[b * (long long)(2 * DC) * DHW + rem] = v;
  }
}

// Per-patch: class softmax over pixels, ctx = probs x pf^T, per-pixel argmax,
// gather local[c][p] = ctx[argmax(p)][c] (bf16).  One 256-thread block / patch.
__global__ __launch_bounds__(256) void k_patch_ctx(
    const float* __restrict__ predpad,          // (B, 32, HW) f32, rows>=19 are zero
    const unsigned short* __restrict__ f2,      // (B, C, H, W) bf16 (after relu6)
    unsigned short* __restrict__ localbf)       // (NP, C, PHW) bf16
{
  __shared__ float pv[DNCLS][DPHW];
  __shared__ float ctxs[DNCLS * DC];
  __shared__ float cmax[DNCLS];
  __shared__ float csum[DNCLS];
  __shared__ int   idxs[DPHW];

  const int n  = blockIdx.x;
  const int pb = n >> 6, ph = (n >> 3) & 7, pw = n & 7;
  const int p  = threadIdx.x;
  const int hw = (ph * DPH + (p >> 4)) * DW + pw * DPW + (p & 15);

  #pragma unroll 1
  for (int k = 0; k < DNCLS; ++k)
    pv[k][p] = predpad[((long long)pb * 32 + k) * DHW + hw];
  __syncthreads();

  const int lane = threadIdx.x & 31;
  const int wid  = threadIdx.x >> 5;
  for (int k = wid; k < DNCLS; k += 8) {
    float m = -3.0e38f;
    for (int e = lane; e < DPHW; e += 32) m = fmaxf(m, pv[k][e]);
    for (int off = 16; off > 0; off >>= 1) m = fmaxf(m, __shfl_xor(m, off, 32));
    float s = 0.f;
    for (int e = lane; e < DPHW; e += 32) s += __expf(pv[k][e] - m);
    for (int off = 16; off > 0; off >>= 1) s += __shfl_xor(s, off, 32);
    if (lane == 0) { cmax[k] = m; csum[k] = s; }
  }
  __syncthreads();

  { // argmax over logits, then overwrite pv with per-class probs
    float best = -3.0e38f; int bi = 0;
    #pragma unroll 1
    for (int k = 0; k < DNCLS; ++k) {
      float lg = pv[k][p];
      if (lg > best) { best = lg; bi = k; }
      pv[k][p] = __expf(lg - cmax[k]) * (1.0f / csum[k]);
    }
    idxs[p] = bi;
  }
  __syncthreads();

  // ctx[k][c] = sum_p probs[k][p] * f[c][p]
  const unsigned short* fbase =
      f2 + (((size_t)pb * DC) * DH + ph * DPH) * DW + pw * DPW;
  for (int o = threadIdx.x; o < DNCLS * DC; o += 256) {
    int k = o >> 9;
    int c = o & 511;
    const unsigned short* fc = fbase + (size_t)c * DH * DW;
    float a = 0.f;
    #pragma unroll 4
    for (int q = 0; q < DPHW; ++q)
      a += pv[k][q] * bf2f(fc[(q >> 4) * DW + (q & 15)]);
    ctxs[o] = a;
  }
  __syncthreads();

  for (int e = threadIdx.x; e < DC * DPHW; e += 256) {
    int c = e >> 8;
    int q = e & 255;
    localbf[((size_t)n * DC + c) * DPHW + q] = f2bf(ctxs[idxs[q] * DC + c]);
  }
}

// row softmax with 1/sqrt(T)=1/16 scale; one wave per 256-wide row
__global__ __launch_bounds__(256) void k_row_softmax_bf16(
    const float* __restrict__ simf, unsigned short* __restrict__ simbf) {
  const int lane = threadIdx.x & 31;
  const int wid  = threadIdx.x >> 5;
  const long long row = (long long)blockIdx.x * 8 + wid;
  const float* src = simf + row * DPHW;
  float v[8];
  float mx = -3.0e38f;
  #pragma unroll
  for (int i = 0; i < 8; ++i) { v[i] = src[lane + i * 32] * 0.0625f; mx = fmaxf(mx, v[i]); }
  #pragma unroll
  for (int off = 16; off > 0; off >>= 1) mx = fmaxf(mx, __shfl_xor(mx, off, 32));
  float s = 0.f;
  #pragma unroll
  for (int i = 0; i < 8; ++i) { v[i] = __expf(v[i] - mx); s += v[i]; }
  #pragma unroll
  for (int off = 16; off > 0; off >>= 1) s += __shfl_xor(s, off, 32);
  float inv = 1.0f / s;
  unsigned short* dst = simbf + row * DPHW;
  #pragma unroll
  for (int i = 0; i < 8; ++i) dst[lane + i * 32] = f2bf(v[i] * inv);
}

// =====================================================================
extern "C" void kernel_launch(void* const* d_in, const int* in_sizes, int n_in,
                              void* d_out, int out_size, void* d_ws, size_t ws_size,
                              hipStream_t stream) {
  const float* feat   = (const float*)d_in[0];
  const float* gc     = (const float*)d_in[1];
  const float* w_proj = (const float*)d_in[2];
  const float* b_proj = (const float*)d_in[3];
  const float* w_dec  = (const float*)d_in[4];
  const float* b_dec  = (const float*)d_in[5];
  const float* q1_w = (const float*)d_in[6];  const float* q1_s = (const float*)d_in[7];  const float* q1_b = (const float*)d_in[8];
  const float* q2_w = (const float*)d_in[9];  const float* q2_s = (const float*)d_in[10]; const float* q2_b = (const float*)d_in[11];
  const float* k1_w = (const float*)d_in[12]; const float* k1_s = (const float*)d_in[13]; const float* k1_b = (const float*)d_in[14];
  const float* k2_w = (const float*)d_in[15]; const float* k2_s = (const float*)d_in[16]; const float* k2_b = (const float*)d_in[17];
  const float* v_w  = (const float*)d_in[18]; const float* v_s  = (const float*)d_in[19]; const float* v_b  = (const float*)d_in[20];
  const float* o_w  = (const float*)d_in[21]; const float* o_s  = (const float*)d_in[22]; const float* o_b  = (const float*)d_in[23];
  const float* cat1_w = (const float*)d_in[24]; const float* cat1_s = (const float*)d_in[25]; const float* cat1_b = (const float*)d_in[26];
  const float* cat2_w = (const float*)d_in[27]; const float* cat2_s = (const float*)d_in[28]; const float* cat2_b = (const float*)d_in[29];

  char* ws = (char*)d_ws;
  size_t off = 0;
  auto alloc = [&](size_t bytes) -> void* {
    off = (off + 255) & ~(size_t)255;
    void* p = ws + off;
    off += bytes;
    return p;
  };

  unsigned short* featbf  = (unsigned short*)alloc((size_t)DB * DC * DHW * 2);       // aliased later as simf
  unsigned short* fprojbf = (unsigned short*)alloc((size_t)DB * DC * DHW * 2);       // aliased later as localbf
  unsigned short* f2bf_   = (unsigned short*)alloc((size_t)DB * DC * DHW * 2);
  unsigned short* xcat    = (unsigned short*)alloc((size_t)DB * 2 * DC * DHW * 2);
  float*          predpad = (float*)alloc((size_t)DB * 32 * DHW * 4);
  unsigned short* wdecbf  = (unsigned short*)alloc((size_t)32 * DC * 2);
  unsigned short* wprojbf = (unsigned short*)alloc((size_t)DC * DC * 2);
  unsigned short* gcbf    = (unsigned short*)alloc((size_t)DB * DC * DC * 2);
  unsigned short* q1wbf   = (unsigned short*)alloc((size_t)DT * DC * 2);
  unsigned short* q2wbf   = (unsigned short*)alloc((size_t)DT * DT * 2);
  unsigned short* k1wbf   = (unsigned short*)alloc((size_t)DT * DC * 2);
  unsigned short* k2wbf   = (unsigned short*)alloc((size_t)DT * DT * 2);
  unsigned short* vwbf    = (unsigned short*)alloc((size_t)DT * DC * 2);
  unsigned short* owbf    = (unsigned short*)alloc((size_t)DC * DT * 2);
  unsigned short* cat1wbf = (unsigned short*)alloc((size_t)DC * 2 * DC * 9 * 2);
  unsigned short* cat2wbf = (unsigned short*)alloc((size_t)DC * DC * 9 * 2);
  unsigned short* q1bf    = (unsigned short*)alloc((size_t)DNP * DT * DPHW * 2);
  unsigned short* k1bf    = (unsigned short*)alloc((size_t)DNP * DT * DPHW * 2);
  unsigned short* qTbf    = (unsigned short*)alloc((size_t)DNP * DPHW * DT * 2);
  unsigned short* kfbf    = (unsigned short*)alloc((size_t)DNP * DT * DPHW * 2);
  unsigned short* vTbf    = (unsigned short*)alloc((size_t)DNP * DPHW * DT * 2);
  unsigned short* simbf   = (unsigned short*)alloc((size_t)DNP * DPHW * DPHW * 2);
  unsigned short* cxtbf   = (unsigned short*)alloc((size_t)DNP * DT * DPHW * 2);
  unsigned short* y1bf    = (unsigned short*)alloc((size_t)DB * DC * DHW * 2);

  float*          simf    = (float*)featbf;   // featbf dead after fproj GEMM
  unsigned short* localbf = fprojbf;          // fprojbf dead after center GEMM

  dim3 blk(256);
  long long total = (long long)DB * DC * DHW;
  k_feat_convert<<<dim3((unsigned)((total + 255) / 256)), blk, 0, stream>>>(feat, featbf, xcat);

  auto cvt = [&](const float* s, unsigned short* d, long long cnt) {
    k_f32_to_bf16<<<dim3((unsigned)((cnt + 255) / 256)), blk, 0, stream>>>(s, d, cnt);
  };
  cvt(w_proj, wprojbf, (long long)DC * DC);
  cvt(gc,     gcbf,    (long long)DB * DC * DC);
  cvt(q1_w,   q1wbf,   (long long)DT * DC);
  cvt(q2_w,   q2wbf,   (long long)DT * DT);
  cvt(k1_w,   k1wbf,   (long long)DT * DC);
  cvt(k2_w,   k2wbf,   (long long)DT * DT);
  cvt(v_w,    vwbf,    (long long)DT * DC);
  cvt(o_w,    owbf,    (long long)DC * DT);
  cvt(cat1_w, cat1wbf, (long long)DC * 2 * DC * 9);
  cvt(cat2_w, cat2wbf, (long long)DC * DC * 9);
  k_convert_pad_rows<<<dim3((32 * DC + 255) / 256), blk, 0, stream>>>(w_dec, wdecbf, DNCLS, 32, DC);

  // 1) pred = w_dec x feat + b_dec   (M padded to 32; pad rows are exact zeros)
  wmma_gemm_kernel<0, 0, 1><<<dim3(DHW / 64, 1, DB), blk, 0, stream>>>(
      wdecbf, 0, featbf, (long long)DC * DHW, predpad, (long long)32 * DHW,
      32, DHW, DC, DNCLS, b_dec, nullptr, 0, 0);

  // 2) fproj = w_proj x feat + b_proj  -> bf16
  wmma_gemm_kernel<0, 1, 1><<<dim3(DHW / 64, DC / 64, DB), blk, 0, stream>>>(
      wprojbf, 0, featbf, (long long)DC * DHW, fprojbf, (long long)DC * DHW,
      DC, DHW, DC, DC, b_proj, nullptr, 0, 0);

  // 3) f2 = relu6(feat + gc[b] x fproj)  -> bf16
  wmma_gemm_kernel<0, 1, 4><<<dim3(DHW / 64, DC / 64, DB), blk, 0, stream>>>(
      gcbf, (long long)DC * DC, fprojbf, (long long)DC * DHW, f2bf_, (long long)DC * DHW,
      DC, DHW, DC, DC, feat, nullptr, (long long)DC * DHW, 0);

  // 4) per-patch softmax/ctx/argmax/gather -> localbf
  k_patch_ctx<<<dim3(DNP), blk, 0, stream>>>(predpad, f2bf_, localbf);

  // 5) q1 = relu(bn(q1_w x pf))   (pf read in-place as patch view of f2)
  wmma_gemm_kernel<1, 1, 2><<<dim3(DPHW / 64, DT / 64, DNP), blk, 0, stream>>>(
      q1wbf, 0, f2bf_, 0, q1bf, (long long)DT * DPHW,
      DT, DPHW, DC, DT, q1_s, q1_b, 0, 0);
  // 6) k1 = relu(bn(k1_w x local))
  wmma_gemm_kernel<0, 1, 2><<<dim3(DPHW / 64, DT / 64, DNP), blk, 0, stream>>>(
      k1wbf, 0, localbf, (long long)DC * DPHW, k1bf, (long long)DT * DPHW,
      DT, DPHW, DC, DT, k1_s, k1_b, 0, 0);
  // 7) v  = relu(bn(v_w x local))  stored transposed (hw, T)
  wmma_gemm_kernel<0, 2, 2><<<dim3(DPHW / 64, DT / 64, DNP), blk, 0, stream>>>(
      vwbf, 0, localbf, (long long)DC * DPHW, vTbf, (long long)DPHW * DT,
      DT, DPHW, DC, DT, v_s, v_b, 0, 0);
  // 8) q2 = relu(bn(q2_w x q1))   stored transposed (hw, T)
  wmma_gemm_kernel<0, 2, 2><<<dim3(DPHW / 64, DT / 64, DNP), blk, 0, stream>>>(
      q2wbf, 0, q1bf, (long long)DT * DPHW, qTbf, (long long)DPHW * DT,
      DT, DPHW, DT, DT, q2_s, q2_b, 0, 0);
  // 9) k2 = relu(bn(k2_w x k1))   (T, hw)
  wmma_gemm_kernel<0, 1, 2><<<dim3(DPHW / 64, DT / 64, DNP), blk, 0, stream>>>(
      k2wbf, 0, k1bf, (long long)DT * DPHW, kfbf, (long long)DT * DPHW,
      DT, DPHW, DT, DT, k2_s, k2_b, 0, 0);

  // 10) sim_raw = qT x kf  (f32, scale applied in softmax)
  wmma_gemm_kernel<0, 0, 0><<<dim3(DPHW / 64, DPHW / 64, DNP), blk, 0, stream>>>(
      qTbf, (long long)DPHW * DT, kfbf, (long long)DT * DPHW, simf, (long long)DPHW * DPHW,
      DPHW, DPHW, DT, DPHW, nullptr, nullptr, 0, 0);
  // 11) sim = softmax(sim_raw / 16)  -> bf16
  k_row_softmax_bf16<<<dim3(DNP * DPHW / 8), blk, 0, stream>>>(simf, simbf);
  // 12) cxt = sim x vT  stored transposed -> (T, hw)
  wmma_gemm_kernel<0, 2, 0><<<dim3(DT / 64, DPHW / 64, DNP), blk, 0, stream>>>(
      simbf, (long long)DPHW * DPHW, vTbf, (long long)DPHW * DT, cxtbf, (long long)DT * DPHW,
      DPHW, DT, DPHW, DPHW, nullptr, nullptr, 0, 0);
  // 13) o = relu(bn(o_w x cxt)) scattered into xcat channels [512, 1024)
  wmma_gemm_kernel<0, 3, 2><<<dim3(DPHW / 64, DC / 64, DNP), blk, 0, stream>>>(
      owbf, 0, cxtbf, (long long)DT * DPHW, xcat, 0,
      DC, DPHW, DT, DC, o_s, o_b, 0, DC);

  // 14) cat1: conv3x3 (1024 -> 512), relu6(bn)  -> bf16
  wmma_conv3x3_kernel<true><<<dim3(DHW / 64, DC / 64, DB), blk, 0, stream>>>(
      cat1wbf, xcat, 2 * DC, cat1_s, cat1_b, y1bf, DC);
  // 15) cat2: conv3x3 (512 -> 512), relu6(bn)  -> f32 output
  wmma_conv3x3_kernel<false><<<dim3(DHW / 64, DC / 64, DB), blk, 0, stream>>>(
      cat2wbf, y1bf, DC, cat2_s, cat2_b, d_out, DC);
}
